// ADTOFFrameRNN_68367289418090
// MI455X (gfx1250) — matrix operand order repaired and checked by reading, not compiled, not run on hardware
//
#include <hip/hip_runtime.h>
#include <hip/hip_bf16.h>

typedef __attribute__((ext_vector_type(16))) _Float16 v16h;
typedef __attribute__((ext_vector_type(8)))  _Float16 v8h;
typedef __attribute__((ext_vector_type(8)))  float    v8f;
typedef __attribute__((ext_vector_type(4)))  float    v4f;
typedef _Float16 half_t;

#define BN_EPS 1e-3f
#define TT 2048
#define XPS 192   // padded xp row stride (floats); cols 180..191 are write-only pad

// Fast nonlinearities: v_exp_f32 + v_rcp_f32 (no IEEE divide sequence).
__device__ __forceinline__ float sigf(float x) {
  return __builtin_amdgcn_rcpf(1.0f + __expf(-x));
}
__device__ __forceinline__ float tanh_fast(float x) {
  return 2.0f * __builtin_amdgcn_rcpf(1.0f + __expf(-2.0f * x)) - 1.0f;
}

// ---------------------------------------------------------------------------
// conv1: Cin=1 -> 32, direct (K=9 too small for WMMA). Fused ReLU + BN.
// x: [4][2048][168] f32 -> out NHWC: [4][2048][168][32] f16
// ---------------------------------------------------------------------------
__global__ void conv1_kernel(const float* __restrict__ x,
                             const float* __restrict__ w, const float* __restrict__ bias,
                             const float* __restrict__ g, const float* __restrict__ be,
                             const float* __restrict__ m, const float* __restrict__ v,
                             half_t* __restrict__ out) {
  int idx = blockIdx.x * blockDim.x + threadIdx.x;  // ((b*T+t)*168+f)*32+c
  int c = idx & 31;
  int f = (idx >> 5) % 168;
  int t = (idx / (32 * 168)) % TT;
  int b = idx / (32 * 168 * TT);
  float acc = bias[c];
#pragma unroll
  for (int ky = 0; ky < 3; ky++) {
    int tt = t + ky - 1;
    if (tt < 0 || tt >= TT) continue;
#pragma unroll
    for (int kx = 0; kx < 3; kx++) {
      int ff = f + kx - 1;
      if (ff < 0 || ff >= 168) continue;
      acc += x[(b * TT + tt) * 168 + ff] * w[(c * 3 + ky) * 3 + kx];
    }
  }
  acc = fmaxf(acc, 0.0f);
  float s = g[c] * rsqrtf(v[c] + BN_EPS);
  out[idx] = (half_t)(s * (acc - m[c]) + be[c]);
}

// ---------------------------------------------------------------------------
// Implicit-GEMM conv3x3 'SAME' via v_wmma_f32_16x16x32_f16, NHWC activations.
// K reduction ordered (tap, ci), ci fastest -> B operand is ONE aligned v16h
// vector load per lane per 32-K chunk (all 16 halves share one (t,f) pos).
// ---------------------------------------------------------------------------
template <int Cin, int Cout, int F>
__global__ void conv_wmma_kernel(const half_t* __restrict__ in,   // [B][T][F][Cin]
                                 const float* __restrict__ w,     // [Cout][Cin][3][3]
                                 const float* __restrict__ bias, const float* __restrict__ gg,
                                 const float* __restrict__ be, const float* __restrict__ mm,
                                 const float* __restrict__ vv,
                                 half_t* __restrict__ out) {      // [B][T][F][Cout]
  constexpr int CT = Cout / 16;
  constexpr int FT = (F + 15) / 16;
  constexpr int nKc = (Cin * 9) / 32;   // Cin is a multiple of 32 -> exact
  int lane = threadIdx.x & 31;
  int wid = blockIdx.x * (blockDim.x >> 5) + (threadIdx.x >> 5);
  int cot = wid % CT;
  int r1 = wid / CT;
  int ft = r1 % FT;
  int r2 = r1 / FT;
  int t = r2 % TT;
  int b = r2 / TT;
  int fpos = ft * 16 + (lane & 15);
  int rowA = cot * 16 + (lane & 15);
  v8f acc = {};
#pragma unroll
  for (int kc = 0; kc < nKc; kc++) {
    v16h a;
    int ka0 = kc * 32 + (lane < 16 ? 0 : 16);
#pragma unroll
    for (int j = 0; j < 16; j++) {
      int k = ka0 + j;
      int ci = k % Cin, tap = k / Cin;              // compile-time shifts
      a[j] = (half_t)w[(rowA * Cin + ci) * 9 + tap];
    }
    int kb0 = kc * 32 + (lane < 16 ? 0 : 16);
    int tap = kb0 / Cin, ci0 = kb0 % Cin;
    int ky = tap / 3, kx = tap % 3;
    int tt = t + ky - 1, ff = fpos + kx - 1;
    v16h bv = {};
    if (fpos < F && tt >= 0 && tt < TT && ff >= 0 && ff < F)
      bv = *(const v16h*)&in[(((long)b * TT + tt) * F + ff) * Cin + ci0];
    acc = __builtin_amdgcn_wmma_f32_16x16x32_f16(false, a, false, bv, (short)0, acc, false, false);
  }
  if (fpos < F) {
    v8h ov;
#pragma unroll
    for (int r = 0; r < 8; r++) {
      int co = cot * 16 + (lane < 16 ? r : 8 + r);
      float val = fmaxf(acc[r] + bias[co], 0.0f);
      float s = gg[co] * rsqrtf(vv[co] + BN_EPS);
      ov[r] = (half_t)(s * (val - mm[co]) + be[co]);
    }
    *(v8h*)&out[(((long)b * TT + t) * F + fpos) * Cout + cot * 16 + (lane < 16 ? 0 : 8)] = ov;
  }
}

// ---------------------------------------------------------------------------
// MaxPool (1,3)/(1,3) over freq, TF-'same' (pad-left 0 both 168->56, 56->19).
// ---------------------------------------------------------------------------
template <int C, int Fin, int Fout>
__global__ void maxpool_kernel(const half_t* __restrict__ in, half_t* __restrict__ out) {
  int idx = blockIdx.x * blockDim.x + threadIdx.x;  // ((b*T+t)*Fout+fo)*C+c
  int c = idx % C;
  int fo = (idx / C) % Fout;
  int t = (idx / (C * Fout)) % TT;
  int b = idx / (C * Fout * TT);
  float best = -1e30f;
#pragma unroll
  for (int k = 0; k < 3; k++) {
    int fi = fo * 3 + k;
    if (fi < Fin) best = fmaxf(best, (float)in[(((long)b * TT + t) * Fin + fi) * C + c]);
  }
  out[idx] = (half_t)best;
}

// ---------------------------------------------------------------------------
// GRU input projection: xp[dir][bt][g] = sum_i x[bt,i]*Wih[g,i] + bih[g].
// A-side pads handled by index CLAMP (B pad columns are guaranteed zero, and
// pad outputs land in xp's write-only pad columns) -> branchless gathers and
// two unconditional v4f stores in the epilogue.
// ---------------------------------------------------------------------------
template <int I, int STRIDE>
__global__ void xp_gemm_kernel(const half_t* __restrict__ xin,   // [8192][STRIDE]
                               const float* __restrict__ wih_f, const float* __restrict__ wih_b,
                               const float* __restrict__ bih_f, const float* __restrict__ bih_b,
                               float* __restrict__ xp) {          // [2][8192][XPS]
  constexpr int nKc = (I + 31) / 32;
  int lane = threadIdx.x & 31;
  int wid = blockIdx.x * (blockDim.x >> 5) + (threadIdx.x >> 5);
  int dir = wid % 2;
  int gt = (wid / 2) % 12;
  int ntile = wid / 24;
  const float* W = dir ? wih_b : wih_f;
  const float* Bi = dir ? bih_b : bih_f;
  int bt = ntile * 16 + (lane & 15);
  int row = min(gt * 16 + (lane & 15), 179);   // clamp: pad rows write xp pad cols
  v8f acc = {};
#pragma unroll 4
  for (int kc = 0; kc < nKc; kc++) {
    v16h a;
    int ka0 = kc * 32 + (lane < 16 ? 0 : 16);
#pragma unroll
    for (int j = 0; j < 16; j++) {
      int k = min(ka0 + j, I - 1);  // clamp: B pad cols are zero
      a[j] = (half_t)W[(long)row * I + k];
    }
    int kb0 = kc * 32 + (lane < 16 ? 0 : 16);
    v16h bv = *(const v16h*)&xin[(long)bt * STRIDE + kb0];  // pad cols are zeroed
    acc = __builtin_amdgcn_wmma_f32_16x16x32_f16(false, a, false, bv, (short)0, acc, false, false);
  }
  int g0 = gt * 16 + (lane < 16 ? 0 : 8);
  v4f o0, o1;
#pragma unroll
  for (int r = 0; r < 4; r++) {
    o0[r] = acc[r] + Bi[min(g0 + r, 179)];
    o1[r] = acc[4 + r] + Bi[min(g0 + 4 + r, 179)];
  }
  float* dst = &xp[((long)dir * 8192 + bt) * XPS + g0];
  *(v4f*)dst = o0;
  *(v4f*)(dst + 4) = o1;
}

// ---------------------------------------------------------------------------
// Zero pad columns (120..127) of both y buffers (xp_gemm loads them).
// ---------------------------------------------------------------------------
__global__ void ypad_kernel(half_t* __restrict__ y0, half_t* __restrict__ y1) {
  int idx = blockIdx.x * blockDim.x + threadIdx.x;  // 65536
  int col = 120 + (idx & 7);
  int rowi = idx >> 3;
  y0[rowi * 128 + col] = (half_t)0.0f;
  y1[rowi * 128 + col] = (half_t)0.0f;
}

// ---------------------------------------------------------------------------
// Recurrent scan. One workgroup per direction (12 waves = 384 thr).
// Whh resident as WMMA A-operands in VGPRs for all 2048 steps; h in LDS.
// Per step: 2 WMMAs/wave -> two ds_store_b128 of hh -> gate math (240 thr)
// with pointer-increment xp/y addressing + prefetch of next step's xp.
// ---------------------------------------------------------------------------
__global__ __launch_bounds__(384) void gru_scan_kernel(
    const float* __restrict__ xp,    // [2][8192][XPS]
    const float* __restrict__ whh_f, const float* __restrict__ whh_b,
    const float* __restrict__ bhh_f, const float* __restrict__ bhh_b,
    half_t* __restrict__ y) {        // [8192][128], cols dir*60..dir*60+59
  int dir = blockIdx.x;
  const float* Whh = dir ? whh_b : whh_f;
  const float* bhh = dir ? bhh_b : bhh_f;
  int tid = threadIdx.x;
  int lane = tid & 31;
  int wv = tid >> 5;  // 0..11: owns g-tile [16*wv, 16*wv+16)

  __shared__ half_t h16[16][64];   // B-operand: rows 0..3 = batch, K padded 64 (zeros)
  __shared__ float hf[4][64];      // f32 master copy of h
  __shared__ float hh[4][192];     // h @ Whh^T; cols 180..191 are pad

  for (int i = tid; i < 16 * 64; i += blockDim.x) (&h16[0][0])[i] = (half_t)0.0f;
  for (int i = tid; i < 4 * 64; i += blockDim.x) (&hf[0][0])[i] = 0.0f;

  // Whh A-operands, index-clamped (pad K cols of h16 are zero; pad g rows of hh
  // are never read) -> branchless gather.
  v16h A0, A1;
  int rowc = min(wv * 16 + (lane & 15), 179);
#pragma unroll
  for (int j = 0; j < 16; j++) {
    int k0 = (lane < 16 ? 0 : 16) + j;
    A0[j] = (half_t)Whh[rowc * 60 + min(k0, 59)];
    A1[j] = (half_t)Whh[rowc * 60 + min(32 + k0, 59)];
  }

  // Gate-phase per-thread state (valid work only for tid < 240)
  int gi = tid < 240 ? tid : 0;
  int b = gi / 60, j = gi - b * 60;
  float bh0 = bhh[j], bh1 = bhh[60 + j], bh2 = bhh[120 + j];
  int t0 = dir ? (TT - 1) : 0;
  const float* xpp = xp + ((long)dir * 8192 + (long)b * TT + t0) * XPS + j;
  half_t* yp = y + ((long)b * TT + t0) * 128 + dir * 60 + j;
  int xstep = dir ? -XPS : XPS;
  int ystep = dir ? -128 : 128;
  __syncthreads();

  for (int s = 0; s < TT; s++) {
    int brow = lane & 15;
    v16h B0 = *(const v16h*)&h16[brow][lane < 16 ? 0 : 16];
    v16h B1 = *(const v16h*)&h16[brow][lane < 16 ? 32 : 48];
    v8f c = {};
    c = __builtin_amdgcn_wmma_f32_16x16x32_f16(false, A0, false, B0, (short)0, c, false, false);
    c = __builtin_amdgcn_wmma_f32_16x16x32_f16(false, A1, false, B1, (short)0, c, false, false);
    if ((lane & 15) < 4) {           // two contiguous b128 stores, no inner guards
      float* dsth = &hh[lane & 15][wv * 16 + (lane < 16 ? 0 : 8)];
      v4f p0 = {c[0], c[1], c[2], c[3]};
      v4f p1 = {c[4], c[5], c[6], c[7]};
      *(v4f*)dsth = p0;
      *(v4f*)(dsth + 4) = p1;
    }
    __syncthreads();
    if (tid < 240) {
      if (s + 1 < TT) {              // warm L1 for next step (global_prefetch_b8)
        __builtin_prefetch(xpp + xstep, 0, 1);
        __builtin_prefetch(xpp + xstep + 60, 0, 1);
        __builtin_prefetch(xpp + xstep + 120, 0, 1);
      }
      float rg = sigf(xpp[0] + bh0 + hh[b][j]);
      float zg = sigf(xpp[60] + bh1 + hh[b][60 + j]);
      float ng = tanh_fast(xpp[120] + rg * (bh2 + hh[b][120 + j]));
      float hnew = (1.0f - zg) * ng + zg * hf[b][j];
      hf[b][j] = hnew;
      h16[b][j] = (half_t)hnew;
      *yp = (half_t)hnew;
      xpp += xstep;
      yp += ystep;
    }
    __syncthreads();
  }
}

// ---------------------------------------------------------------------------
// Output head: sigmoid(y @ out_w^T + out_b) -> [B*T][5] f32
// ---------------------------------------------------------------------------
__global__ void dense_kernel(const half_t* __restrict__ yv, const float* __restrict__ ow,
                             const float* __restrict__ ob, float* __restrict__ out) {
  int idx = blockIdx.x * blockDim.x + threadIdx.x;
  int o = idx % 5;
  long bt = idx / 5;
  float acc = ob[o];
  for (int i = 0; i < 120; i++) acc += (float)yv[bt * 128 + i] * ow[o * 120 + i];
  out[idx] = sigf(acc);
}

extern "C" void kernel_launch(void* const* d_in, const int* in_sizes, int n_in,
                              void* d_out, int out_size, void* d_ws, size_t ws_size,
                              hipStream_t stream) {
  (void)in_sizes; (void)n_in; (void)out_size; (void)ws_size;
  const float* x = (const float*)d_in[0];
  const float *cw[5], *cb[5], *cg[5], *cbe[5], *cm[5], *cv[5];
  int p = 1;
  for (int i = 1; i <= 4; i++) {
    cw[i] = (const float*)d_in[p++];  cb[i] = (const float*)d_in[p++];
    cg[i] = (const float*)d_in[p++];  cbe[i] = (const float*)d_in[p++];
    cm[i] = (const float*)d_in[p++];  cv[i] = (const float*)d_in[p++];
  }
  const float *wih[3][2], *whh[3][2], *bih[3][2], *bhh[3][2];
  for (int l = 0; l < 3; l++)
    for (int d = 0; d < 2; d++) {
      wih[l][d] = (const float*)d_in[p++]; whh[l][d] = (const float*)d_in[p++];
      bih[l][d] = (const float*)d_in[p++]; bhh[l][d] = (const float*)d_in[p++];
    }
  const float* ow = (const float*)d_in[p++];
  const float* ob = (const float*)d_in[p++];

  // Workspace arena (liveness-based reuse; peak ~192 MB):
  //   bufA: act1 -> pool1 -> act4 -> (xp, y0, y1)
  //   bufB: act2 -> act3 -> pool2
  char* ws = (char*)d_ws;
  half_t* bufA = (half_t*)ws;                               // 96 MB
  half_t* bufB = (half_t*)(ws + ((size_t)96 << 20));        // 96 MB
  half_t* act1 = bufA;       half_t* act2 = bufB;
  half_t* pool1 = bufA;      half_t* act3 = bufB;
  half_t* act4 = bufA;       half_t* pool2 = bufB;
  float* xp = (float*)ws;                                   // 12.6 MB (bufA free post-pool2)
  half_t* y0 = (half_t*)(ws + ((size_t)16 << 20));          // 2 MB  [8192][128]
  half_t* y1 = (half_t*)(ws + ((size_t)24 << 20));          // 2 MB

  conv1_kernel<<<(4 * 32 * TT * 168) / 256, 256, 0, stream>>>(
      x, cw[1], cb[1], cg[1], cbe[1], cm[1], cv[1], act1);
  conv_wmma_kernel<32, 32, 168><<<(4 * TT * 11 * 2) / 8, 256, 0, stream>>>(
      act1, cw[2], cb[2], cg[2], cbe[2], cm[2], cv[2], act2);
  maxpool_kernel<32, 168, 56><<<(4 * 32 * TT * 56) / 256, 256, 0, stream>>>(act2, pool1);
  conv_wmma_kernel<32, 64, 56><<<(4 * TT * 4 * 4) / 8, 256, 0, stream>>>(
      pool1, cw[3], cb[3], cg[3], cbe[3], cm[3], cv[3], act3);
  conv_wmma_kernel<64, 64, 56><<<(4 * TT * 4 * 4) / 8, 256, 0, stream>>>(
      act3, cw[4], cb[4], cg[4], cbe[4], cm[4], cv[4], act4);
  maxpool_kernel<64, 56, 19><<<(4 * 64 * TT * 19) / 256, 256, 0, stream>>>(act4, pool2);

  ypad_kernel<<<65536 / 256, 256, 0, stream>>>(y0, y1);

  // 3 stacked BiGRU layers (scan: 2 persistent workgroups, fwd/bwd concurrent)
  xp_gemm_kernel<1216, 1216><<<1536, 256, 0, stream>>>(
      pool2, wih[0][0], wih[0][1], bih[0][0], bih[0][1], xp);
  gru_scan_kernel<<<2, 384, 0, stream>>>(xp, whh[0][0], whh[0][1], bhh[0][0], bhh[0][1], y0);
  xp_gemm_kernel<120, 128><<<1536, 256, 0, stream>>>(
      y0, wih[1][0], wih[1][1], bih[1][0], bih[1][1], xp);
  gru_scan_kernel<<<2, 384, 0, stream>>>(xp, whh[1][0], whh[1][1], bhh[1][0], bhh[1][1], y1);
  xp_gemm_kernel<120, 128><<<1536, 256, 0, stream>>>(
      y1, wih[2][0], wih[2][1], bih[2][0], bih[2][1], xp);
  gru_scan_kernel<<<2, 384, 0, stream>>>(xp, whh[2][0], whh[2][1], bhh[2][0], bhh[2][1], y0);

  dense_kernel<<<160, 256, 0, stream>>>(y0, ow, ob, (float*)d_out);
}
